// MultiHeadAttention_25615184953550
// MI455X (gfx1250) — matrix operand order repaired and checked by reading, not compiled, not run on hardware
//
#include <hip/hip_runtime.h>
#include <hip/hip_bf16.h>

// ---------------------------------------------------------------------------
// MHA forward for B=2, S=2048, H=16, D=64, E=1024 on gfx1250 (MI455X).
// fp32 -> bf16 once, all matmuls via V_WMMA_F32_16X16X32_BF16 (fp32 accum),
// flash-attention with online softmax (DPP row reductions), fp32 output.
// ---------------------------------------------------------------------------

typedef __bf16 bf16;
typedef __attribute__((ext_vector_type(8)))  __bf16 v8bf;
typedef __attribute__((ext_vector_type(16))) __bf16 v16bf;
typedef __attribute__((ext_vector_type(8)))  float  v8f;

constexpr int B_ = 2, S_ = 2048, H_ = 16, D_ = 64, E_ = 1024;
constexpr int MROWS = B_ * S_;
constexpr float SCALE = 0.125f;         // 1/sqrt(64)
constexpr int LP  = 40;                 // flash P-tile LDS row stride (bf16)
constexpr int LKT = 72;                 // gemm LDS row stride for K-tile=64

__device__ __forceinline__ v16bf join16(const bf16* p0, const bf16* p1) {
  v8bf lo = *(const v8bf*)p0;
  v8bf hi = *(const v8bf*)p1;
  return __builtin_shufflevector(lo, hi, 0,1,2,3,4,5,6,7,8,9,10,11,12,13,14,15);
}

__device__ __forceinline__ v8f wmma_bf16(v16bf a, v16bf b, v8f c) {
  return __builtin_amdgcn_wmma_f32_16x16x32_bf16(false, a, false, b,
                                                 (short)0, c, false, false);
}

// DPP16 lane permute (immediate control): quad_perm xor1/xor2, half/row mirror
template <int CTRL>
__device__ __forceinline__ float dpp_mov(float v) {
  return __builtin_bit_cast(float,
      __builtin_amdgcn_update_dpp(0, __builtin_bit_cast(int, v),
                                  CTRL, 0xF, 0xF, true));
}
__device__ __forceinline__ float row_max16(float v) {
  v = fmaxf(v, dpp_mov<0x0B1>(v));   // quad_perm {1,0,3,2}  (xor 1)
  v = fmaxf(v, dpp_mov<0x04E>(v));   // quad_perm {2,3,0,1}  (xor 2)
  v = fmaxf(v, dpp_mov<0x141>(v));   // row_half_mirror      (xor 7)
  v = fmaxf(v, dpp_mov<0x140>(v));   // row_mirror           (xor 15)
  return v;
}
__device__ __forceinline__ float row_sum16(float v) {
  v += dpp_mov<0x0B1>(v);
  v += dpp_mov<0x04E>(v);
  v += dpp_mov<0x141>(v);
  v += dpp_mov<0x140>(v);
  return v;
}

// ---------------------------------------------------------------------------
// Conversions / transposes
// ---------------------------------------------------------------------------
__global__ __launch_bounds__(256) void k_f32_to_bf16(const float* __restrict__ in,
                                                     bf16* __restrict__ out, int n) {
  int i = blockIdx.x * 256 + threadIdx.x;
  if (i < n) out[i] = (bf16)in[i];
}

// W [E x E] row-major -> Wt [N][K] = W^T in bf16 (coalesced writes)
__global__ __launch_bounds__(256) void k_transpose_w(const float* __restrict__ W,
                                                     bf16* __restrict__ Wt) {
  int i = blockIdx.x * 256 + threadIdx.x;   // i = n*E + k
  int n = i >> 10;
  int k = i & (E_ - 1);
  Wt[i] = (bf16)W[k * E_ + n];
}

// V [B,S,E] bf16 -> Vt [B,H,D,S] bf16 (coalesced writes)
__global__ __launch_bounds__(256) void k_transpose_v(const bf16* __restrict__ Vb,
                                                     bf16* __restrict__ Vt) {
  int i = blockIdx.x * 256 + threadIdx.x;   // i = ((b*H+h)*D+d)*S + s
  int s = i & (S_ - 1);
  int r = i >> 11;
  int d = r & (D_ - 1);
  r >>= 6;
  int h = r & (H_ - 1);
  int b = r >> 4;
  Vt[i] = Vb[((size_t)(b * S_ + s)) * E_ + h * D_ + d];
}

// ---------------------------------------------------------------------------
// GEMM: C[M,N] = A[M,K] * Bt[N,K]^T, bf16 in, fp32 accumulate.
// 128x128 block tile, 8 waves (16x128 strip each), K staged 64-wide in LDS
// so each barrier pair feeds 16 WMMAs per wave.
// ---------------------------------------------------------------------------
template <bool OUT_BF16>
__global__ __launch_bounds__(256) void k_gemm_bf16(const bf16* __restrict__ A,
                                                   const bf16* __restrict__ Bt,
                                                   void* __restrict__ Cout,
                                                   int M, int N, int Kd) {
  __shared__ bf16 Al[128 * LKT];
  __shared__ bf16 Bl[128 * LKT];

  const int tid  = threadIdx.x;
  const int wave = tid >> 5;
  const int lane = tid & 31;
  const int l15  = lane & 15;
  const int half = (lane >> 4) & 1;        // 0: lanes 0-15, 1: lanes 16-31
  const int m0   = blockIdx.y * 128;
  const int n0   = blockIdx.x * 128;

  v8f acc[8] = {};

  const int lr = tid >> 1;                 // 0..127: tile row this thread loads
  const int lc = (tid & 1) * 32;           // 0 or 32: k-offset (32 elems each)

  for (int k0 = 0; k0 < Kd; k0 += 64) {
    {
      const bf16* ag = A  + (size_t)(m0 + lr) * Kd + k0 + lc;
      const bf16* bg = Bt + (size_t)(n0 + lr) * Kd + k0 + lc;
#pragma unroll
      for (int q = 0; q < 4; ++q) {
        v8bf av = *(const v8bf*)(ag + q * 8);
        v8bf bv = *(const v8bf*)(bg + q * 8);
        *(v8bf*)&Al[lr * LKT + lc + q * 8] = av;
        *(v8bf*)&Bl[lr * LKT + lc + q * 8] = bv;
      }
    }
    __syncthreads();

    const int am  = wave * 16 + l15;
    const int kb8 = half * 8;
#pragma unroll
    for (int kk = 0; kk < 64; kk += 32) {
      v16bf afrag = join16(&Al[am * LKT + kk + kb8],
                           &Al[am * LKT + kk + kb8 + 16]);
#pragma unroll
      for (int nt = 0; nt < 8; ++nt) {
        const int bn = nt * 16 + l15;
        v16bf bfrag = join16(&Bl[bn * LKT + kk + kb8],
                             &Bl[bn * LKT + kk + kb8 + 16]);
        acc[nt] = wmma_bf16(afrag, bfrag, acc[nt]);
      }
    }
    __syncthreads();
  }

  const int rowbase = m0 + wave * 16 + half * 8;
#pragma unroll
  for (int nt = 0; nt < 8; ++nt) {
    const int col = n0 + nt * 16 + l15;
#pragma unroll
    for (int r = 0; r < 8; ++r) {
      const size_t idx = (size_t)(rowbase + r) * N + col;
      if constexpr (OUT_BF16) ((bf16*)Cout)[idx] = (bf16)acc[nt][r];
      else                    ((float*)Cout)[idx] = acc[nt][r];
    }
  }
}

// ---------------------------------------------------------------------------
// Flash attention (causal). One wave = 16 query rows; key blocks of 32.
// Exactly one key block per wave straddles the diagonal, so the main loop
// runs unmasked and only the final block pays the compare/select cost.
// ---------------------------------------------------------------------------
__global__ __launch_bounds__(256) void k_flash_attn(const bf16* __restrict__ Q,
                                                    const bf16* __restrict__ K,
                                                    const bf16* __restrict__ Vt,
                                                    bf16* __restrict__ O) {
  __shared__ bf16 plds[8 * 16 * LP];       // per-wave 16x32 P tile (padded)

  const int tid  = threadIdx.x;
  const int wave = tid >> 5;
  const int lane = tid & 31;
  const int l15  = lane & 15;
  const int half = (lane >> 4) & 1;
  const int b    = blockIdx.y / H_;
  const int h    = blockIdx.y % H_;
  const int q0   = blockIdx.x * 128 + wave * 16;

  const bf16* Qh = Q  + (size_t)b * S_ * E_ + h * D_;
  const bf16* Kh = K  + (size_t)b * S_ * E_ + h * D_;
  const bf16* Vh = Vt + ((size_t)(b * H_ + h)) * D_ * S_;
  bf16* lds = &plds[wave * 16 * LP];

  // Q A-fragments (d = 0..31, 32..63), resident for the whole kernel
  v16bf aq[2];
  {
    const int kb8 = half * 8;
    const bf16* qrow = Qh + (size_t)(q0 + l15) * E_;
#pragma unroll
    for (int f = 0; f < 2; ++f)
      aq[f] = join16(qrow + f * 32 + kb8, qrow + f * 32 + kb8 + 16);
  }

  v8f acc[4] = {};                         // 16 rows x 64 d, fp32
  float mrow[8], lrow[8];
#pragma unroll
  for (int r = 0; r < 8; ++r) { mrow[r] = -3.0e38f; lrow[r] = 0.0f; }

  auto process_block = [&](int k0, bool masked) __attribute__((always_inline)) {
    // --- scores S = Q * K^T (two 16-key subtiles, 2 k-steps over d) ---
    v8f sc[2] = {};
#pragma unroll
    for (int sub = 0; sub < 2; ++sub) {
      const int key = k0 + sub * 16 + l15;         // B-frag: lane = key column
      const bf16* krow = Kh + (size_t)key * E_;
#pragma unroll
      for (int f = 0; f < 2; ++f) {
        v16bf bk = join16(krow + f * 32 + half * 8, krow + f * 32 + half * 8 + 16);
        sc[sub] = wmma_bf16(aq[f], bk, sc[sub]);
      }
    }

    // --- scale (+ causal mask on the diagonal block), row max via DPP ---
    float rmax[8];
#pragma unroll
    for (int r = 0; r < 8; ++r) {
      const int qg = q0 + r + half * 8;
#pragma unroll
      for (int sub = 0; sub < 2; ++sub) {
        float v = sc[sub][r] * SCALE;
        if (masked) {
          const int kg = k0 + sub * 16 + l15;
          v = (kg <= qg) ? v : -3.0e38f;
        }
        sc[sub][r] = v;
      }
      rmax[r] = row_max16(fmaxf(sc[0][r], sc[1][r]));
    }

    // --- online softmax update ---
    float alpha[8];
#pragma unroll
    for (int r = 0; r < 8; ++r) {
      float mnew = fmaxf(mrow[r], rmax[r]);
      alpha[r] = __expf(mrow[r] - mnew);
      mrow[r] = mnew;
    }
#pragma unroll
    for (int r = 0; r < 8; ++r) {
      const int m = r + half * 8;
      float rs = 0.0f;
#pragma unroll
      for (int sub = 0; sub < 2; ++sub) {
        float p = __expf(sc[sub][r] - mrow[r]);
        rs += p;
        lds[m * LP + sub * 16 + l15] = (bf16)p;   // stash P for A-layout reload
      }
      lrow[r] = lrow[r] * alpha[r] + row_sum16(rs);
#pragma unroll
      for (int t = 0; t < 4; ++t) acc[t][r] *= alpha[r];
    }

    asm volatile("s_wait_dscnt 0" ::: "memory");   // P stores -> P loads

    // --- O += P * V (P: 16x32 A-frag from LDS; V: contiguous from Vt) ---
    const int kb8 = half * 8;
    v16bf pa = join16(&lds[l15 * LP + kb8], &lds[l15 * LP + kb8 + 16]);
#pragma unroll
    for (int t = 0; t < 4; ++t) {
      const int d = t * 16 + l15;                  // B-frag: lane = d column
      v16bf bv = join16(Vh + (size_t)d * S_ + k0 + kb8, Vh + (size_t)d * S_ + k0 + kb8 + 16);
      acc[t] = wmma_bf16(pa, bv, acc[t]);
    }
  };

  const int nfull = (q0 + 1) >> 5;         // blocks fully below the diagonal
  for (int kb = 0; kb < nfull; ++kb) process_block(kb * 32, false);
  process_block(nfull * 32, true);         // the single diagonal block

  // --- normalize + store (bf16, [B,S,E] head slice) ---
  bf16* Oh = O + (size_t)b * S_ * E_ + h * D_;
#pragma unroll
  for (int r = 0; r < 8; ++r) {
    const int row = q0 + r + half * 8;
    const float inv = 1.0f / lrow[r];
#pragma unroll
    for (int t = 0; t < 4; ++t)
      Oh[(size_t)row * E_ + t * 16 + l15] = (bf16)(acc[t][r] * inv);
  }
}

// ---------------------------------------------------------------------------
// Host orchestration
// ---------------------------------------------------------------------------
extern "C" void kernel_launch(void* const* d_in, const int* in_sizes, int n_in,
                              void* d_out, int out_size, void* d_ws, size_t ws_size,
                              hipStream_t stream) {
  (void)in_sizes; (void)n_in; (void)out_size; (void)ws_size;
  const float* x  = (const float*)d_in[0];
  const float* Wq = (const float*)d_in[1];
  const float* Wk = (const float*)d_in[2];
  const float* Wv = (const float*)d_in[3];
  const float* Wo = (const float*)d_in[4];

  char* ws = (char*)d_ws;
  const size_t MB = 1u << 20;
  bf16* xb  = (bf16*)(ws);              // 8 MB  [4096,1024]
  bf16* Wqt = (bf16*)(ws +  8 * MB);    // 2 MB  [N,K]
  bf16* Wkt = (bf16*)(ws + 10 * MB);
  bf16* Wvt = (bf16*)(ws + 12 * MB);
  bf16* Wot = (bf16*)(ws + 14 * MB);
  bf16* Qb  = (bf16*)(ws + 16 * MB);    // 8 MB  [B,S,E]
  bf16* Kb  = (bf16*)(ws + 24 * MB);
  bf16* Vb  = (bf16*)(ws + 32 * MB);
  bf16* Vtb = (bf16*)(ws + 40 * MB);    // 8 MB  [B,H,D,S]
  bf16* Ob  = (bf16*)(ws + 48 * MB);    // 8 MB  [B,S,E]

  const int nx = MROWS * E_;
  const int nw = E_ * E_;

  k_f32_to_bf16<<<nx / 256, 256, 0, stream>>>(x, xb, nx);
  k_transpose_w<<<nw / 256, 256, 0, stream>>>(Wq, Wqt);
  k_transpose_w<<<nw / 256, 256, 0, stream>>>(Wk, Wkt);
  k_transpose_w<<<nw / 256, 256, 0, stream>>>(Wv, Wvt);
  k_transpose_w<<<nw / 256, 256, 0, stream>>>(Wo, Wot);

  dim3 gg(E_ / 128, MROWS / 128);       // (8, 32)
  k_gemm_bf16<true><<<gg, 256, 0, stream>>>(xb, Wqt, Qb, MROWS, E_, E_);
  k_gemm_bf16<true><<<gg, 256, 0, stream>>>(xb, Wkt, Kb, MROWS, E_, E_);
  k_gemm_bf16<true><<<gg, 256, 0, stream>>>(xb, Wvt, Vb, MROWS, E_, E_);

  k_transpose_v<<<(B_ * H_ * D_ * S_) / 256, 256, 0, stream>>>(Vb, Vtb);

  dim3 fg(S_ / 128, B_ * H_);           // (16, 32)
  k_flash_attn<<<fg, 256, 0, stream>>>(Qb, Kb, Vtb, Ob);

  k_gemm_bf16<false><<<gg, 256, 0, stream>>>(Ob, Wot, d_out, MROWS, E_, E_);
}
